// AttentionBlock_47888885351038
// MI455X (gfx1250) — matrix operand order repaired and checked by reading, not compiled, not run on hardware
//
#include <hip/hip_runtime.h>
#include <hip/hip_bf16.h>
#include <stdint.h>

// ---------------------------------------------------------------------------
// AttentionBlock on MI455X (gfx1250): GroupNorm -> QKV GEMM -> softmax(QtK) -> PV -> proj + residual
// All large GEMMs run on v_wmma_f32_16x16x32_bf16 (fp32 accumulate).
// Per-wave tile: 64(M) x 32(N)  (4x2 WMMA tiles); block = 8 waves -> 256(M) x 64(N).
// ---------------------------------------------------------------------------

#define BB   4
#define CC   512
#define SSZ  4096           // H*W
#define OO   1536           // 3*C
#define CG   16             // channels per group (512/32)
#define SCALE 0.044194173824159216f   // 512^-0.5

#define MT 4                // 16-row tiles per wave (M)
#define NT 2                // 16-col tiles per wave (N)
#define BLK_M 256           // 4 waves in M * MT*16
#define BLK_N 64            // 2 waves in N * NT*16

typedef __attribute__((ext_vector_type(16))) __bf16 v16bf;
typedef __attribute__((ext_vector_type(8)))  float  v8f;

union FragBF { v16bf bf; uint32_t u[8]; };
union FragF  { v8f  f;  float    x[8]; };

__device__ __forceinline__ uint16_t f2bf(float f) {
  uint32_t u = __float_as_uint(f);
  u += 0x7FFFu + ((u >> 16) & 1u);          // round-to-nearest-even
  return (uint16_t)(u >> 16);
}
__device__ __forceinline__ float bf2f(uint16_t h) {
  return __uint_as_float((uint32_t)h << 16);
}

// A fragment: 16x32 bf16, rows m, stored row-major [M][K] (pairs contiguous in K)
__device__ __forceinline__ void load_a_frag(FragBF& fa, const uint16_t* __restrict__ A,
                                            int lda, int m0, int k0, int lane) {
  const int m = m0 + (lane & 15);
  const int h = lane >> 4;
  const uint16_t* base = A + (size_t)m * lda + k0;
#pragma unroll
  for (int v = 0; v < 8; ++v) {
    const int k = ((v & 4) << 2) + (h << 3) + ((v & 3) << 1);   // (v<4?0:16)+8h+2(v&3)
    fa.u[v] = *(const uint32_t*)(base + k);
  }
}
// B fragment: 32x16 bf16, cols n, source stored [N][K] (pairs contiguous in K)
__device__ __forceinline__ void load_b_frag(FragBF& fb, const uint16_t* __restrict__ Bm,
                                            int ldb, int n0, int k0, int lane) {
  const int n = n0 + (lane & 15);
  const int h = lane >> 4;
  const uint16_t* base = Bm + (size_t)n * ldb + k0;
#pragma unroll
  for (int v = 0; v < 8; ++v) fb.u[v] = *(const uint32_t*)(base + ((v & 4) << 2) + (h << 3) + ((v & 3) << 1));
}

__device__ __forceinline__ v8f wmma_bf16(const FragBF& a, const FragBF& b, v8f c) {
  return __builtin_amdgcn_wmma_f32_16x16x32_bf16(false, a.bf, false, b.bf,
                                                 (short)0, c, false, false);
}

// shared GEMM core: acc[MT][NT] over K, A row-major [M][K], B as [N][K]
#define GEMM_CORE(Am, lda, Bm, ldb, KDIM)                                     \
  FragF acc[MT][NT];                                                          \
  _Pragma("unroll")                                                           \
  for (int i = 0; i < MT; ++i)                                                \
    _Pragma("unroll")                                                         \
    for (int j = 0; j < NT; ++j)                                              \
      _Pragma("unroll")                                                       \
      for (int e = 0; e < 8; ++e) acc[i][j].x[e] = 0.f;                       \
  for (int k0 = 0; k0 < (KDIM); k0 += 32) {                                   \
    FragBF a[MT], bb[NT];                                                     \
    _Pragma("unroll")                                                         \
    for (int i = 0; i < MT; ++i) load_a_frag(a[i], Am, lda, m_base + i * 16, k0, lane); \
    _Pragma("unroll")                                                         \
    for (int j = 0; j < NT; ++j) load_b_frag(bb[j], Bm, ldb, n_base + j * 16, k0, lane); \
    _Pragma("unroll")                                                         \
    for (int i = 0; i < MT; ++i)                                              \
      _Pragma("unroll")                                                       \
      for (int j = 0; j < NT; ++j)                                            \
        acc[i][j].f = wmma_bf16(a[i], bb[j], acc[i][j].f);                    \
  }

// ---------------------------------------------------------------------------
// 1) GroupNorm statistics: one block per (b, group); 16 ch * 4096 = 65536 elems
// ---------------------------------------------------------------------------
__global__ __launch_bounds__(256)
void gn_stats_kernel(const float* __restrict__ x, float* __restrict__ stats) {
  const int bg = blockIdx.x;                 // 0..127
  const float* xp = x + (size_t)bg * CG * SSZ;
  float s = 0.f, ss = 0.f;
  for (int i = threadIdx.x; i < CG * SSZ; i += 256) {
    float v = xp[i]; s += v; ss += v * v;
  }
  __shared__ float ls[256], lss[256];
  ls[threadIdx.x] = s; lss[threadIdx.x] = ss;
  __syncthreads();
  for (int st = 128; st > 0; st >>= 1) {
    if (threadIdx.x < st) { ls[threadIdx.x] += ls[threadIdx.x + st];
                            lss[threadIdx.x] += lss[threadIdx.x + st]; }
    __syncthreads();
  }
  if (threadIdx.x == 0) {
    const float inv_n = 1.0f / (float)(CG * SSZ);
    float mu  = ls[0] * inv_n;
    float var = lss[0] * inv_n - mu * mu;
    stats[2 * bg]     = mu;
    stats[2 * bg + 1] = rsqrtf(var + 1e-5f);
  }
}

// ---------------------------------------------------------------------------
// 2) fp32 -> bf16 weight conversion
// ---------------------------------------------------------------------------
__global__ __launch_bounds__(256)
void cvt_bf16_kernel(const float* __restrict__ in, uint16_t* __restrict__ out, int n) {
  int i = blockIdx.x * 256 + threadIdx.x;
  if (i < n) out[i] = f2bf(in[i]);
}

// ---------------------------------------------------------------------------
// 3) normalize + affine + transpose: xnT[b][s][c] (bf16)
// ---------------------------------------------------------------------------
__global__ __launch_bounds__(256)
void gn_apply_kernel(const float* __restrict__ x, const float* __restrict__ stats,
                     const float* __restrict__ gw, const float* __restrict__ gb,
                     uint16_t* __restrict__ xnT) {
  size_t idx = (size_t)blockIdx.x * 256 + threadIdx.x;   // b,c,s with s fastest
  const size_t CS = (size_t)CC * SSZ;
  int b = (int)(idx / CS);
  size_t rem = idx - (size_t)b * CS;
  int c = (int)(rem / SSZ);
  int s = (int)(rem - (size_t)c * SSZ);
  int g = c >> 4;                                        // c / CG
  float mu  = stats[2 * (b * 32 + g)];
  float inv = stats[2 * (b * 32 + g) + 1];
  float xn = (x[idx] - mu) * inv * gw[c] + gb[c];
  xnT[((size_t)b * SSZ + s) * CC + c] = f2bf(xn);
}

// ---------------------------------------------------------------------------
// 4) QKV GEMM: W(1536x512) x XnT(b)(4096x512 as [n][k]) -> qT/kT ([s][c]) , v ([c][s])
// ---------------------------------------------------------------------------
__global__ __launch_bounds__(256)
void qkv_gemm_kernel(const uint16_t* __restrict__ qkvw, const uint16_t* __restrict__ xnT,
                     const float* __restrict__ qkvb,
                     uint16_t* __restrict__ qT, uint16_t* __restrict__ kT,
                     uint16_t* __restrict__ vbuf) {
  const int lane = threadIdx.x & 31, w = threadIdx.x >> 5;
  const int m_base = blockIdx.x * BLK_M + (w >> 1) * (MT * 16);
  const int n_base = blockIdx.y * BLK_N + (w & 1) * (NT * 16);
  const int b = blockIdx.z;
  const uint16_t* Bm = xnT + (size_t)b * SSZ * CC;

  GEMM_CORE(qkvw, CC, Bm, CC, CC)

  const int h = lane >> 4, nl = lane & 15;
#pragma unroll
  for (int ti = 0; ti < MT; ++ti)
#pragma unroll
    for (int tj = 0; tj < NT; ++tj) {
      const int s  = n_base + tj * 16 + nl;
      const int o0 = m_base + ti * 16 + h * 8;    // 8 consecutive output channels
      if (o0 < 1024) {                            // q or k -> transposed packed store
        uint16_t* dst = (o0 < 512) ? qT : kT;
        const int ch0 = o0 & 511;
        uint32_t pk[4];
#pragma unroll
        for (int p = 0; p < 4; ++p) {
          float f0 = acc[ti][tj].x[2 * p]     + qkvb[o0 + 2 * p];
          float f1 = acc[ti][tj].x[2 * p + 1] + qkvb[o0 + 2 * p + 1];
          pk[p] = (uint32_t)f2bf(f0) | ((uint32_t)f2bf(f1) << 16);
        }
        uint32_t* outp = (uint32_t*)(dst + ((size_t)b * SSZ + s) * CC + ch0);
        outp[0] = pk[0]; outp[1] = pk[1]; outp[2] = pk[2]; outp[3] = pk[3];
      } else {                                    // v -> [c][s]
        const int c0 = o0 - 1024;
#pragma unroll
        for (int v = 0; v < 8; ++v) {
          float f = acc[ti][tj].x[v] + qkvb[o0 + v];
          vbuf[((size_t)b * CC + c0 + v) * SSZ + s] = f2bf(f);
        }
      }
    }
}

// ---------------------------------------------------------------------------
// 5) scores: S[b][i][j] = scale * sum_c qT[i][c] * kT[j][c]   (M=N=4096, K=512)
// ---------------------------------------------------------------------------
__global__ __launch_bounds__(256)
void scores_kernel(const uint16_t* __restrict__ qT, const uint16_t* __restrict__ kT,
                   uint16_t* __restrict__ attnw) {
  const int lane = threadIdx.x & 31, w = threadIdx.x >> 5;
  const int m_base = blockIdx.x * BLK_M + (w >> 1) * (MT * 16);
  const int n_base = blockIdx.y * BLK_N + (w & 1) * (NT * 16);
  const int b = blockIdx.z;
  const uint16_t* Am = qT + (size_t)b * SSZ * CC;
  const uint16_t* Bm = kT + (size_t)b * SSZ * CC;

  GEMM_CORE(Am, CC, Bm, CC, CC)

  const int h = lane >> 4, nl = lane & 15;
  uint16_t* dst = attnw + (size_t)b * SSZ * SSZ;
#pragma unroll
  for (int ti = 0; ti < MT; ++ti)
#pragma unroll
    for (int tj = 0; tj < NT; ++tj) {
      const int j = n_base + tj * 16 + nl;
#pragma unroll
      for (int v = 0; v < 8; ++v) {
        const int i = m_base + ti * 16 + h * 8 + v;
        dst[(size_t)i * SSZ + j] = f2bf(acc[ti][tj].x[v] * SCALE);
      }
    }
}

// ---------------------------------------------------------------------------
// 6) row softmax: one wave per row; stores exp(s-max) in bf16, 1/sum to rinv
// ---------------------------------------------------------------------------
__global__ __launch_bounds__(256)
void softmax_kernel(uint16_t* __restrict__ attnw, float* __restrict__ rinv) {
  const int lane = threadIdx.x & 31;
  const int row  = blockIdx.x * 8 + (threadIdx.x >> 5);   // 0 .. B*S-1
  uint16_t* rp = attnw + (size_t)row * SSZ;

  float mx = -3.0e38f;
  for (int j = lane; j < SSZ; j += 32) mx = fmaxf(mx, bf2f(rp[j]));
#pragma unroll
  for (int d = 16; d > 0; d >>= 1) mx = fmaxf(mx, __shfl_xor(mx, d, 32));

  float sum = 0.f;
  for (int j = lane; j < SSZ; j += 32) {
    float p = __expf(bf2f(rp[j]) - mx);
    uint16_t pb = f2bf(p);
    rp[j] = pb;
    sum += bf2f(pb);                 // sum what we actually stored
  }
#pragma unroll
  for (int d = 16; d > 0; d >>= 1) sum += __shfl_xor(sum, d, 32);

  if (lane == 0) rinv[row] = 1.0f / sum;
}

// ---------------------------------------------------------------------------
// 7) PV: outT[b][i][c] = rinv[i] * sum_j P[i][j] * v[c][j]   (M=4096,N=512,K=4096)
// ---------------------------------------------------------------------------
__global__ __launch_bounds__(256)
void attnout_kernel(const uint16_t* __restrict__ attnw, const uint16_t* __restrict__ vbuf,
                    const float* __restrict__ rinv, uint16_t* __restrict__ outT) {
  const int lane = threadIdx.x & 31, w = threadIdx.x >> 5;
  const int m_base = blockIdx.x * BLK_M + (w >> 1) * (MT * 16);
  const int n_base = blockIdx.y * BLK_N + (w & 1) * (NT * 16);
  const int b = blockIdx.z;
  const uint16_t* Am = attnw + (size_t)b * SSZ * SSZ;   // [i][j]
  const uint16_t* Bm = vbuf  + (size_t)b * CC  * SSZ;   // [c][j] == [n][k]

  GEMM_CORE(Am, SSZ, Bm, SSZ, SSZ)

  const int h = lane >> 4, nl = lane & 15;
#pragma unroll
  for (int ti = 0; ti < MT; ++ti)
#pragma unroll
    for (int tj = 0; tj < NT; ++tj) {
      const int c = n_base + tj * 16 + nl;
#pragma unroll
      for (int v = 0; v < 8; ++v) {
        const int i = m_base + ti * 16 + h * 8 + v;
        const float rv = rinv[b * SSZ + i];
        outT[((size_t)b * SSZ + i) * CC + c] = f2bf(acc[ti][tj].x[v] * rv);
      }
    }
}

// ---------------------------------------------------------------------------
// 8) proj + bias + residual: out[b][o][s] = x + Wp x outT^T + proj_b
// ---------------------------------------------------------------------------
__global__ __launch_bounds__(256)
void proj_kernel(const uint16_t* __restrict__ projw, const uint16_t* __restrict__ outT,
                 const float* __restrict__ projb, const float* __restrict__ xin,
                 float* __restrict__ outp) {
  const int lane = threadIdx.x & 31, w = threadIdx.x >> 5;
  const int m_base = blockIdx.x * BLK_M + (w >> 1) * (MT * 16);
  const int n_base = blockIdx.y * BLK_N + (w & 1) * (NT * 16);
  const int b = blockIdx.z;
  const uint16_t* Bm = outT + (size_t)b * SSZ * CC;     // [s][c] == [n][k]

  GEMM_CORE(projw, CC, Bm, CC, CC)

  const int h = lane >> 4, nl = lane & 15;
#pragma unroll
  for (int ti = 0; ti < MT; ++ti)
#pragma unroll
    for (int tj = 0; tj < NT; ++tj) {
      const int s = n_base + tj * 16 + nl;
#pragma unroll
      for (int v = 0; v < 8; ++v) {
        const int o = m_base + ti * 16 + h * 8 + v;
        const size_t idx = ((size_t)b * CC + o) * SSZ + s;
        outp[idx] = xin[idx] + acc[ti][tj].x[v] + projb[o];
      }
    }
}

// ---------------------------------------------------------------------------
extern "C" void kernel_launch(void* const* d_in, const int* in_sizes, int n_in,
                              void* d_out, int out_size, void* d_ws, size_t ws_size,
                              hipStream_t stream) {
  const float* x      = (const float*)d_in[0];
  const float* gn_w   = (const float*)d_in[1];
  const float* gn_b   = (const float*)d_in[2];
  const float* qkv_w  = (const float*)d_in[3];
  const float* qkv_b  = (const float*)d_in[4];
  const float* proj_w = (const float*)d_in[5];
  const float* proj_b = (const float*)d_in[6];
  float* out = (float*)d_out;
  (void)in_sizes; (void)n_in; (void)out_size; (void)ws_size;

  // workspace carve-up (~210 MB; attn in bf16 fits the 192MB L2)
  char* base = (char*)d_ws;
  size_t off = 0;
  auto take = [&](size_t bytes) { size_t o = off; off = (off + bytes + 255) & ~(size_t)255; return o; };
  float*    stats  = (float*)   (base + take(BB * 32 * 2 * sizeof(float)));
  float*    rinv   = (float*)   (base + take((size_t)BB * SSZ * sizeof(float)));
  uint16_t* qkvwbf = (uint16_t*)(base + take((size_t)OO * CC * 2));
  uint16_t* projwbf= (uint16_t*)(base + take((size_t)CC * CC * 2));
  uint16_t* xnT    = (uint16_t*)(base + take((size_t)BB * SSZ * CC * 2));
  uint16_t* qT     = (uint16_t*)(base + take((size_t)BB * SSZ * CC * 2));
  uint16_t* kT     = (uint16_t*)(base + take((size_t)BB * SSZ * CC * 2));
  uint16_t* vbuf   = (uint16_t*)(base + take((size_t)BB * CC * SSZ * 2));
  uint16_t* outT   = (uint16_t*)(base + take((size_t)BB * SSZ * CC * 2));
  uint16_t* attnw  = (uint16_t*)(base + take((size_t)BB * SSZ * SSZ * 2));

  // 1) group-norm stats
  gn_stats_kernel<<<dim3(BB * 32), dim3(256), 0, stream>>>(x, stats);
  // 2) weight conversion
  cvt_bf16_kernel<<<dim3((OO * CC + 255) / 256), dim3(256), 0, stream>>>(qkv_w, qkvwbf, OO * CC);
  cvt_bf16_kernel<<<dim3((CC * CC + 255) / 256), dim3(256), 0, stream>>>(proj_w, projwbf, CC * CC);
  // 3) normalize + transpose to xnT[b][s][c]
  gn_apply_kernel<<<dim3((unsigned)(((size_t)BB * CC * SSZ) / 256)), dim3(256), 0, stream>>>(
      x, stats, gn_w, gn_b, xnT);
  // 4) QKV GEMM  (M=1536)
  qkv_gemm_kernel<<<dim3(OO / BLK_M, SSZ / BLK_N, BB), dim3(256), 0, stream>>>(
      qkvwbf, xnT, qkv_b, qT, kT, vbuf);
  // 5) attention scores (M=N=4096)
  scores_kernel<<<dim3(SSZ / BLK_M, SSZ / BLK_N, BB), dim3(256), 0, stream>>>(qT, kT, attnw);
  // 6) softmax rows
  softmax_kernel<<<dim3((BB * SSZ) / 8), dim3(256), 0, stream>>>(attnw, rinv);
  // 7) P x V^T  (M=4096, N=512)
  attnout_kernel<<<dim3(SSZ / BLK_M, CC / BLK_N, BB), dim3(256), 0, stream>>>(
      attnw, vbuf, rinv, outT);
  // 8) proj + residual (M=512)
  proj_kernel<<<dim3(CC / BLK_M, SSZ / BLK_N, BB), dim3(256), 0, stream>>>(
      projwbf, outT, proj_b, x, out);
}
